// CoordinateDecoder_79568564126192
// MI455X (gfx1250) — matrix operand order repaired and checked by reading, not compiled, not run on hardware
//
#include <hip/hip_runtime.h>

typedef __attribute__((ext_vector_type(16))) _Float16 v16h;
typedef __attribute__((ext_vector_type(2)))  _Float16 h2;
typedef __attribute__((ext_vector_type(8)))  float    v8f;
typedef unsigned int u32;

#define S160  0.07905694150420949f   // 1/sqrt(160)
#define S192  0.07216878364870323f   // 1/sqrt(192)
#define SQRT2F 1.41421356237309515f

// ---------------------------------------------------------------------------
// Pre-pass 1: pack weights into WMMA A-operand fragments (f16, contiguous-k A layout).
// Fragment order: L1 t=0..3 (1 K-chunk) | L2 (t,c) 8 | L3 (t,c) 8 | L4 c=0,1.
// Each fragment = 32 lanes x 16 halfs, lane-major (lane loads 32B = v16h).
// A 16-bit 16x32 layout: lane l (l2=l&15,hi=l>>4) holds row M=l2,
//   half h -> k = hi*16 + h.
// Layers 2/3/4 absorb the sqrt(2) lrelu gain of their input activation:
//   lrelu(v)*sqrt2 @ W  ==  max(v, 0.2v) @ (sqrt2*W).
// ---------------------------------------------------------------------------
__global__ __launch_bounds__(256) void cd_pack_weights(
    const float* __restrict__ w1, const float* __restrict__ w2,
    const float* __restrict__ w3, const float* __restrict__ w4,
    unsigned short* __restrict__ frags) {
  int gid = blockIdx.x * 256 + threadIdx.x;
  if (gid >= 22 * 512) return;
  int f    = gid >> 9;
  int lane = (gid >> 4) & 31;
  int h    = gid & 15;
  int l2 = lane & 15, hi = lane >> 4;
  int kl = hi * 16 + h;
  const float S192R2 = S192 * SQRT2F;
  float val;
  if (f < 4) {                       // layer1: W1[:, 0:32] (emb part), no gain
    int j = f * 16 + l2;
    val = w1[j * 160 + kl] * S160;
  } else if (f < 12) {               // layer2: W2[:, 0:64] * sqrt2
    int g = f - 4; int t = g >> 1, cch = g & 1;
    int j = t * 16 + l2; int k = cch * 32 + kl;
    val = w2[j * 192 + k] * S192R2;
  } else if (f < 20) {               // layer3: W3[:, 0:64] * sqrt2
    int g = f - 12; int t = g >> 1, cch = g & 1;
    int j = t * 16 + l2; int k = cch * 32 + kl;
    val = w3[j * 192 + k] * S192R2;
  } else {                           // layer4: W4[0:4, 0:64] * sqrt2, rows>=4 zero
    int cch = f - 20; int j = l2; int k = cch * 32 + kl;
    val = (j < 4) ? w4[j * 192 + k] * S192R2 : 0.0f;
  }
  _Float16 hv = (_Float16)val;
  frags[gid] = __builtin_bit_cast(unsigned short, hv);
}

// ---------------------------------------------------------------------------
// Pre-pass 2: fold the per-batch cond contribution into fp32 biases (exact).
// bias layout (floats): [0:512) L1 (8b x 64j) | [512:1024) L2 | [1024:1536) L3
//                       | [1536:1568) L4 (8b x 4j)
// ---------------------------------------------------------------------------
__global__ __launch_bounds__(64) void cd_bias(
    const float* __restrict__ cnd,
    const float* __restrict__ w1, const float* __restrict__ b1,
    const float* __restrict__ w2, const float* __restrict__ b2,
    const float* __restrict__ w3, const float* __restrict__ b3,
    const float* __restrict__ w4, const float* __restrict__ b4,
    float* __restrict__ bias) {
  int b = blockIdx.x, j = threadIdx.x;   // 8 blocks x 64 threads
  const float* cb = cnd + b * 128;
  float s1 = 0.f, s2 = 0.f, s3 = 0.f;
  for (int k = 0; k < 128; ++k) {
    float cv = cb[k];
    s1 += w1[j * 160 + 32 + k] * cv;
    s2 += w2[j * 192 + 64 + k] * cv;
    s3 += w3[j * 192 + 64 + k] * cv;
  }
  bias[0 * 512 + b * 64 + j] = b1[j] + s1 * S160;
  bias[1 * 512 + b * 64 + j] = b2[j] + s2 * S192;
  bias[2 * 512 + b * 64 + j] = b3[j] + s3 * S192;
  if (j < 4) {
    float s4 = 0.f;
    for (int k = 0; k < 128; ++k) s4 += w4[j * 192 + 64 + k] * cb[k];
    bias[3 * 512 + b * 4 + j] = b4[j] + s4 * S192;
  }
}

// ---------------------------------------------------------------------------
// D-tiles -> fused f16 activation -> next-layer B-operands.
// Convert to packed f16 (v_cvt_pk_f16_f32), apply max(v, 0.2v) (one
// v_pk_mul_f16 + one v_pk_max_num_f16; the sqrt2 gain lives in the next
// layer's weights), then half-wave exchange on packed u32 words.
// D (f32 16x16): lane l = col N(pixel)=l&15, element r = row M = r + 8*(l>>4).
// B (f16 32x16): lane l = col N(pixel)=l&15, half h -> K = (l>>4)*16 + h.
// B chunk c pulls rows of tile T = 2c + hi: rows 0..7 from the hi=0 lane of
// this pixel, rows 8..15 from the hi=1 lane.
// ---------------------------------------------------------------------------
__device__ __forceinline__ void cd_make_b(const v8f acc[4], int hi,
                                          v16h& bh0, v16h& bh1) {
  const _Float16 B = (_Float16)0.2f;
  u32 pk[4][4];
#pragma unroll
  for (int T = 0; T < 4; ++T)
#pragma unroll
    for (int i = 0; i < 4; ++i) {
      h2 v; v[0] = (_Float16)acc[T][2 * i]; v[1] = (_Float16)acc[T][2 * i + 1];
      v = __builtin_elementwise_max(v, v * B);   // lrelu (gain in weights)
      pk[T][i] = __builtin_bit_cast(u32, v);
    }

  u32 s[4][4];
#pragma unroll
  for (int T = 0; T < 4; ++T)
#pragma unroll
    for (int i = 0; i < 4; ++i)
      s[T][i] = (u32)__shfl_xor((int)pk[T][i], 16, 32);

  union { v16h h; u32 u[8]; } b0, b1;
#pragma unroll
  for (int i = 0; i < 4; ++i) {
    b0.u[i]     = hi ? s[1][i]  : pk[0][i];   // tile 2c+hi rows 0..7
    b0.u[4 + i] = hi ? pk[1][i] : s[0][i];    // tile 2c+hi rows 8..15
    b1.u[i]     = hi ? s[3][i]  : pk[2][i];
    b1.u[4 + i] = hi ? pk[3][i] : s[2][i];
  }
  bh0 = b0.h; bh1 = b1.h;
}

// ---------------------------------------------------------------------------
// Main: each wave handles 16 pixels; layers computed as D = W * X^T so the
// final tile delivers out[pixel][0..3] per lane -> coalesced float4 stores.
// ---------------------------------------------------------------------------
__global__ __launch_bounds__(256) void cd_main(
    const float* __restrict__ coords, const float* __restrict__ fu_p,
    const unsigned short* __restrict__ frags, const float* __restrict__ bias,
    float* __restrict__ out, int npix) {
  const int lane = threadIdx.x & 31;
  const int wave = threadIdx.x >> 5;
  const int l2 = lane & 15, hi = lane >> 4;
  const int p  = (blockIdx.x * 8 + wave) * 16 + l2;
  const int pc = (p < npix) ? p : (npix - 1);
  const int b  = pc >> 16;                 // 256*256 pixels per batch image
  const float x = coords[2 * pc + 0];
  const float y = coords[2 * pc + 1];
  const float fu = fu_p[0];

  // ---- layer-1 B operand: 32 masked sinusoidal dims for this lane's pixel.
  // Lane covers k = hi*16 + h, so freq fq = hi*4 + (h>>2) and t = h&3 (both
  // reduced to per-lane constants): arg = coord * 0.1*2^fq,
  // wgt = clip(fu*9 - fq - 1, 0, 1). Native v_sin/v_cos: |arg| <~ 130 rad is
  // in range, and f16 conversion swamps the approximation error.
  v16h bh0, bh1;
  {
    const float fb = fu * 9.0f - (float)(hi * 4) - 1.0f;  // wgt = clip(fb - hc)
    const float sc = hi ? 1.6f : 0.1f;                    // 0.1 * 2^(4*hi)
    v16h be;
#pragma unroll
    for (int h = 0; h < 16; ++h) {
      const int hc = h >> 2;                              // compile-time
      float wgt = fminf(fmaxf(fb - (float)hc, 0.0f), 1.0f);
      float arg = (((h & 1) ? y : x)) * (sc * (float)(1 << hc));
      float v = ((h & 2) == 0) ? __sinf(arg) : __cosf(arg);
      be[h] = (_Float16)(v * wgt);
    }
    bh0 = be;
  }

  v8f acc[4];
  // ---- layer 1: 4 row-tiles of W1emb (K=32); lrelu deferred to cd_make_b --
  {
    const float* bb = bias + b * 64;
#pragma unroll
    for (int t = 0; t < 4; ++t) {
      v8f a;
#pragma unroll
      for (int r = 0; r < 8; ++r) a[r] = bb[t * 16 + hi * 8 + r];
      v16h w = *(const v16h*)(frags + (size_t)t * 512 + lane * 16);
      acc[t] = __builtin_amdgcn_wmma_f32_16x16x32_f16(false, w, false, bh0,
                                                      (short)0, a, false, false);
    }
  }

  // ---- layers 2 and 3: 4 row-tiles x 2 K-chunks (K=64) --------------------
#pragma unroll
  for (int L = 0; L < 2; ++L) {
    cd_make_b(acc, hi, bh0, bh1);          // applies layer-(L+1) lrelu in f16
    const unsigned short* fb = frags + (size_t)(4 + L * 8) * 512;
    const float* bb = bias + (size_t)(1 + L) * 512 + b * 64;
#pragma unroll
    for (int t = 0; t < 4; ++t) {
      v8f a;
#pragma unroll
      for (int r = 0; r < 8; ++r) a[r] = bb[t * 16 + hi * 8 + r];
      v16h w0 = *(const v16h*)(fb + (size_t)(t * 2 + 0) * 512 + lane * 16);
      v16h w1 = *(const v16h*)(fb + (size_t)(t * 2 + 1) * 512 + lane * 16);
      a = __builtin_amdgcn_wmma_f32_16x16x32_f16(false, w0, false, bh0,
                                                 (short)0, a, false, false);
      acc[t] = __builtin_amdgcn_wmma_f32_16x16x32_f16(false, w1, false, bh1,
                                                      (short)0, a, false, false);
    }
  }

  // ---- layer 4: one row-tile (only rows 0..3 live), no activation ---------
  cd_make_b(acc, hi, bh0, bh1);            // applies layer-3 lrelu in f16
  v8f o;
  {
    const float* bb = bias + 3 * 512 + b * 4;
#pragma unroll
    for (int r = 0; r < 8; ++r) {
      int j = hi * 8 + r;
      o[r] = (j < 4) ? bb[j] : 0.0f;
    }
    v16h w0 = *(const v16h*)(frags + (size_t)20 * 512 + lane * 16);
    v16h w1 = *(const v16h*)(frags + (size_t)21 * 512 + lane * 16);
    o = __builtin_amdgcn_wmma_f32_16x16x32_f16(false, w0, false, bh0,
                                               (short)0, o, false, false);
    o = __builtin_amdgcn_wmma_f32_16x16x32_f16(false, w1, false, bh1,
                                               (short)0, o, false, false);
  }

  // ---- radius mask + store: lanes 0..15 hold out[pixel l2][j] in elems 0..3
  float rad  = sqrtf(x * x + y * y);
  float mask = 1.0f - tanhf(fmaxf(rad - 1.0f, 0.0f));
  if (hi == 0 && p < npix) {
    float4 ov = make_float4(o[0] * mask, o[1] * mask, o[2] * mask, o[3] * mask);
    *(float4*)(out + (size_t)p * 4) = ov;
  }
}

// ---------------------------------------------------------------------------
extern "C" void kernel_launch(void* const* d_in, const int* in_sizes, int n_in,
                              void* d_out, int out_size, void* d_ws, size_t ws_size,
                              hipStream_t stream) {
  const float* coords = (const float*)d_in[0];
  const float* cnd    = (const float*)d_in[1];
  const float* fu     = (const float*)d_in[2];
  const float* w1     = (const float*)d_in[3];
  const float* b1     = (const float*)d_in[4];
  const float* w2     = (const float*)d_in[5];
  const float* b2     = (const float*)d_in[6];
  const float* w3     = (const float*)d_in[7];
  const float* b3     = (const float*)d_in[8];
  const float* w4     = (const float*)d_in[9];
  const float* b4     = (const float*)d_in[10];

  unsigned short* frags = (unsigned short*)d_ws;            // 22*512 u16 = 22528 B
  float* bias = (float*)((char*)d_ws + 22528);              // 1568 floats

  cd_pack_weights<<<44, 256, 0, stream>>>(w1, w2, w3, w4, frags);
  cd_bias<<<8, 64, 0, stream>>>(cnd, w1, b1, w2, b2, w3, b3, w4, b4, bias);

  int npix   = in_sizes[0] / 2;                             // 524288
  int blocks = (npix + 127) / 128;                          // 16 px/wave * 8 waves
  cd_main<<<blocks, 256, 0, stream>>>(coords, fu, frags, bias, (float*)d_out, npix);
}
